// TransConvEncoderModule_8555574854348
// MI455X (gfx1250) — compile-verified, matrix operand order checked by reading, not run on hardware
//
#include <hip/hip_runtime.h>
#include <hip/hip_bf16.h>
#include <math.h>

typedef __attribute__((ext_vector_type(16))) _Float16 v16h;
typedef __attribute__((ext_vector_type(8)))  float    v8f;
typedef __attribute__((ext_vector_type(2)))  float    v2f;

#if defined(__has_builtin)
#if __has_builtin(__builtin_amdgcn_wmma_f32_16x16x4_f32)
#define HAVE_WMMA_F32 1
#endif
#endif

namespace {
constexpr int B_   = 8;
constexpr int CIN0 = 512;
constexpr int H_   = 32;
constexpr int W_   = 96;
constexpr int C_   = 64;
constexpr int CQK_ = 16;
constexpr int N_   = H_ * W_;            // 3072
constexpr int KSEL = 614;                // int(3072*0.2)
constexpr int KPAD = 640;
constexpr int QT_  = (KSEL + 15) / 16;   // 39
constexpr int CK   = 128;                // key chunk for fused attention
constexpr int NCH  = N_ / CK;            // 24
constexpr float BN_EPS = 1e-5f;

// ---- workspace layout (in floats) ----
constexpr size_t SZ_POS = (size_t)C_ * N_;
constexpr size_t SZ_BCN = (size_t)B_ * C_ * N_;
constexpr size_t SZ_Q   = (size_t)B_ * N_ * CQK_;
constexpr size_t OFF_POS  = 0;
constexpr size_t OFF_X    = OFF_POS + SZ_POS;
constexpr size_t OFF_Q    = OFF_X   + SZ_BCN;
constexpr size_t OFF_K    = OFF_Q   + SZ_Q;
constexpr size_t OFF_V    = OFF_K   + SZ_Q;
constexpr size_t OFF_O16  = OFF_V   + SZ_BCN;
constexpr size_t OFF_O32  = OFF_O16 + SZ_BCN;
constexpr size_t OFF_IMP  = OFF_O32 + SZ_BCN;
constexpr size_t OFF_IDX  = OFF_IMP + (size_t)B_ * N_;      // ints live here
constexpr size_t OFF_MASK = OFF_IDX + (size_t)B_ * KPAD;
constexpr size_t OFF_B16  = OFF_MASK + (size_t)B_ * N_;
constexpr size_t OFF_B32  = OFF_B16 + SZ_BCN;
constexpr size_t OFF_S16  = OFF_B32 + SZ_BCN;
constexpr size_t OFF_S32  = OFF_S16 + SZ_BCN;
} // namespace

// -------------------- positional embedding --------------------
__global__ __launch_bounds__(256) void pos_kernel(float* __restrict__ pos) {
  int i = blockIdx.x * blockDim.x + threadIdx.x;
  if (i >= C_ * N_) return;
  int c = i / N_;
  int n = i % N_;
  int h = n / W_;
  int w = n % W_;
  int j, coord;
  if (c < 32) { j = c;      coord = h + 1; }
  else        { j = c - 32; coord = w + 1; }
  float dim_t = __powf(10000.0f, 2.0f * (float)(j >> 1) / 32.0f);
  float a = (float)coord / dim_t;
  pos[i] = (j & 1) ? __cosf(a) : __sinf(a);
}

// -------------------- conv3x3 + BN + ReLU (+pos, +f16 round) --------------------
// Implicit-GEMM tiling: one block = 16 pixels x 64 output channels,
// 4 waves, each wave owns a 16-channel tile. Reduction axis = cin*9.
// Uses V_WMMA_F32_16X16X4_F32 when the builtin is available; scalar fallback
// uses the identical launch geometry so host/device stay consistent.
__global__ __launch_bounds__(128) void conv_bn_kernel(
    const float* __restrict__ in, int cin,
    const float* __restrict__ wgt, const float* __restrict__ bn,
    const float* __restrict__ pos, float* __restrict__ out, int round16) {
  const int bx    = blockIdx.x;
  const int b     = bx / (N_ / 16);
  const int nbase = (bx % (N_ / 16)) * 16;
  const int t     = threadIdx.x;
  const int KTOT  = cin * 9;

#if defined(HAVE_WMMA_F32)
  const int lane = t & 31;
  const int wid  = t >> 5;
  const int m    = lane & 15;     // A row (co-in-tile) and B col (pixel-in-tile)
  const int hiL  = lane >> 4;
  const int cob  = wid * 16;
  const int p    = nbase + m;
  const int h    = p / W_;
  const int w    = p % W_;
  v8f acc = {};
  for (int ks = 0; ks < KTOT; ks += 4) {
    v2f a, bb;
#pragma unroll
    for (int r = 0; r < 2; ++r) {
      int kk = ks + r + 2 * hiL;                       // K index 0..3 within step
      a[r] = wgt[(size_t)(cob + m) * KTOT + kk];       // weight[co][ci*9+r9]
      int ci = kk / 9;
      int r9 = kk - ci * 9;
      int dh = r9 / 3 - 1;
      int dw = r9 - (r9 / 3) * 3 - 1;
      int hh = h + dh, ww = w + dw;
      bb[r] = (hh >= 0 && hh < H_ && ww >= 0 && ww < W_)
                  ? in[((size_t)(b * cin + ci) * H_ + hh) * W_ + ww]
                  : 0.f;
    }
    acc = __builtin_amdgcn_wmma_f32_16x16x4_f32(false, a, false, bb,
                                                (short)0, acc, false, false);
  }
#pragma unroll
  for (int r = 0; r < 8; ++r) {
    int co = cob + r + 8 * hiL;
    float s = bn[co], b2 = bn[C_ + co], mm = bn[2 * C_ + co], vv = bn[3 * C_ + co];
    float inv = s * rsqrtf(vv + BN_EPS);
    float val = acc[r] * inv + (b2 - mm * inv);
    val = fmaxf(val, 0.f);
    if (pos) val += pos[(size_t)co * N_ + p];
    if (round16) val = (float)(_Float16)val;
    out[((size_t)(b * C_ + co)) * N_ + p] = val;
  }
#else
  // scalar fallback: 1024 outputs per block, 8 per thread
  for (int o = t; o < 16 * C_; o += 128) {
    int co = o >> 4;
    int p  = nbase + (o & 15);
    int h  = p / W_;
    int w  = p % W_;
    float acc = 0.f;
    const float* wg = wgt + (size_t)co * KTOT;
    for (int ci = 0; ci < cin; ++ci) {
      const float* ip = in + ((size_t)(b * cin + ci) * H_) * W_;
      __builtin_prefetch(ip + (size_t)H_ * W_, 0, 1);
      const float* wc = wg + ci * 9;
#pragma unroll
      for (int dh = 0; dh < 3; ++dh) {
        int hh = h + dh - 1;
        if (hh < 0 || hh >= H_) continue;
#pragma unroll
        for (int dw = 0; dw < 3; ++dw) {
          int ww = w + dw - 1;
          if (ww < 0 || ww >= W_) continue;
          acc += ip[hh * W_ + ww] * wc[dh * 3 + dw];
        }
      }
    }
    float s = bn[co], b2 = bn[C_ + co], mm = bn[2 * C_ + co], vv = bn[3 * C_ + co];
    float inv = s * rsqrtf(vv + BN_EPS);
    float val = acc * inv + (b2 - mm * inv);
    val = fmaxf(val, 0.f);
    if (pos) val += pos[(size_t)co * N_ + p];
    if (round16) val = (float)(_Float16)val;
    out[((size_t)(b * C_ + co)) * N_ + p] = val;
  }
#endif
}

// -------------------- QKV projection --------------------
__global__ __launch_bounds__(256) void qkv_kernel(
    const float* __restrict__ x,
    const float* __restrict__ qw, const float* __restrict__ qb,
    const float* __restrict__ kw, const float* __restrict__ kb,
    const float* __restrict__ vw, const float* __restrict__ vb,
    float* __restrict__ q, float* __restrict__ k, float* __restrict__ v) {
  int i = blockIdx.x * blockDim.x + threadIdx.x;
  if (i >= B_ * N_ * 96) return;
  int o  = i % 96;
  int bn = i / 96;
  int n  = bn % N_;
  int b  = bn / N_;
  const float* xp = x + ((size_t)b * C_) * N_ + n;
  if (o < CQK_) {
    float acc = qb[o];
    const float* wr = qw + o * C_;
    for (int c = 0; c < C_; ++c) acc += wr[c] * xp[(size_t)c * N_];
    q[((size_t)b * N_ + n) * CQK_ + o] = acc;
  } else if (o < 2 * CQK_) {
    int oo = o - CQK_;
    float acc = kb[oo];
    const float* wr = kw + oo * C_;
    for (int c = 0; c < C_; ++c) acc += wr[c] * xp[(size_t)c * N_];
    k[((size_t)b * CQK_ + oo) * N_ + n] = acc;
  } else {
    int oo = o - 2 * CQK_;
    float acc = vb[oo];
    const float* wr = vw + oo * C_;
    for (int c = 0; c < C_; ++c) acc += wr[c] * xp[(size_t)c * N_];
    v[((size_t)b * C_ + oo) * N_ + n] = acc;
  }
}

__global__ __launch_bounds__(256) void zero_kernel(float* __restrict__ p, int n) {
  int i = blockIdx.x * blockDim.x + threadIdx.x;
  if (i < n) p[i] = 0.f;
}

__global__ __launch_bounds__(256) void copy_kernel(float* __restrict__ d,
                                                   const float* __restrict__ s, int n) {
  int i = blockIdx.x * blockDim.x + threadIdx.x;
  if (i < n) d[i] = s[i];
}

// -------------------- async global->LDS staging of a k chunk --------------------
// Stages k[b][0:16][ch*CK : ch*CK+CK] (fp32) into a 16xCK LDS tile using the
// CDNA5 async DMA path (ASYNCcnt). 512 x 16B segments, 128 threads -> 4 each.
static __device__ __forceinline__ void stage_k_chunk(
    const float* __restrict__ k, int b, int ch, float* __restrict__ dst, int t) {
  const float* src = k + ((size_t)b * CQK_) * N_ + ch * CK;
#pragma unroll
  for (int s = 0; s < 4; ++s) {
    int seg = t + s * 128;            // 0..511
    int row = seg >> 5;               // 32 segments (512B) per row
    int c16 = seg & 31;
    const float* gp = src + (size_t)row * N_ + c16 * 4;
    float*       lp = dst + row * CK + c16 * 4;
    asm volatile("global_load_async_to_lds_b128 %0, %1, off"
                 :: "v"((unsigned)(unsigned long long)lp),
                    "v"((unsigned long long)gp)
                 : "memory");
  }
}
#define ASYNC_WAIT() asm volatile("s_wait_asynccnt 0x0" ::: "memory")

// -------------------- fused fp16 attention (3-pass streaming softmax, WMMA) --------------------
__global__ __launch_bounds__(128) void attn_kernel(
    const float* __restrict__ q, const float* __restrict__ k, const float* __restrict__ v,
    float* __restrict__ o16, float* __restrict__ imp) {
  __shared__ __align__(16) float KT[2][CQK_][CK];   // double-buffered k chunk
  __shared__ float    S[16][CK];
  __shared__ _Float16 ATT[16][CK];
  __shared__ float    rmax[16], rsum[16];

  const int bq    = blockIdx.x;
  const int b     = bq / (N_ / 16);
  const int qbase = (bq % (N_ / 16)) * 16;
  const int t     = threadIdx.x;
  const int lane  = t & 31;
  const int wid   = t >> 5;          // 4 waves (wave32)
  const int mrow  = lane & 15;
  const int hiL   = lane >> 4;

  // A fragment: 16 queries x 32 (c_qk=16 zero-padded). ISA 16-bit A 16x32 layout.
  v16h afrag;
#pragma unroll
  for (int h = 0; h < 16; ++h) {
    int K = (h & 7) + 8 * (hiL + 2 * (h >> 3));
    afrag[h] = (K < CQK_) ? (_Float16)q[((size_t)(b * N_ + qbase + mrow)) * CQK_ + K]
                          : (_Float16)0.f;
  }

  if (t < 16) { rmax[t] = -1e30f; rsum[t] = 0.f; }
  __syncthreads();

  // energy from an LDS-staged k tile
  auto energy_from = [&](const float (*kt)[CK]) {
#pragma unroll
    for (int s = 0; s < 2; ++s) {
      int ko = (2 * wid + s) * 16;
      v16h bfrag;
#pragma unroll
      for (int h = 0; h < 16; ++h) {
        int K = (h & 7) + 8 * (hiL + 2 * (h >> 3));
        bfrag[h] = (K < CQK_) ? (_Float16)kt[K][ko + mrow] : (_Float16)0.f;
      }
      v8f c = {};
      c = __builtin_amdgcn_wmma_f32_16x16x32_f16(false, afrag, false, bfrag,
                                                 (short)0, c, false, false);
#pragma unroll
      for (int r = 0; r < 8; ++r) S[r + 8 * hiL][ko + mrow] = c[r];
    }
  };

  // ---- pass 1: row max ----
  stage_k_chunk(k, b, 0, &KT[0][0][0], t);
  for (int ch = 0; ch < NCH; ++ch) {
    ASYNC_WAIT();
    __syncthreads();                                   // KT[ch&1] ready everywhere
    if (ch + 1 < NCH) stage_k_chunk(k, b, ch + 1, &KT[(ch + 1) & 1][0][0], t);
    energy_from(KT[ch & 1]);
    __syncthreads();
    if (t < 16) {
      float m = rmax[t];
      for (int j = 0; j < CK; ++j) m = fmaxf(m, S[t][j]);
      rmax[t] = m;
    }
    __syncthreads();
  }
  // ---- pass 2: row sum of exp ----
  stage_k_chunk(k, b, 0, &KT[0][0][0], t);
  for (int ch = 0; ch < NCH; ++ch) {
    ASYNC_WAIT();
    __syncthreads();
    if (ch + 1 < NCH) stage_k_chunk(k, b, ch + 1, &KT[(ch + 1) & 1][0][0], t);
    energy_from(KT[ch & 1]);
    __syncthreads();
    if (t < 16) {
      float s = rsum[t], m = rmax[t];
      for (int j = 0; j < CK; ++j) s += __expf(S[t][j] - m);
      rsum[t] = s;
    }
    __syncthreads();
  }
  // ---- pass 3: att (fp16) -> out16 GEMM + importance column sums ----
  v8f oacc = {};   // wave wid owns channel group [16*wid, 16*wid+16)
  stage_k_chunk(k, b, 0, &KT[0][0][0], t);
  for (int ch = 0; ch < NCH; ++ch) {
    ASYNC_WAIT();
    __syncthreads();
    if (ch + 1 < NCH) stage_k_chunk(k, b, ch + 1, &KT[(ch + 1) & 1][0][0], t);
    energy_from(KT[ch & 1]);
    __syncthreads();
    {
      int j = t;  // 128 columns
      float cs = 0.f;
#pragma unroll
      for (int i = 0; i < 16; ++i) {
        float a = __expf(S[i][j] - rmax[i]) / rsum[i];
        ATT[i][j] = (_Float16)a;
        cs += (float)ATT[i][j];
      }
      atomicAdd(&imp[(size_t)b * N_ + ch * CK + j], cs);
    }
    __syncthreads();
#pragma unroll
    for (int kk = 0; kk < CK; kk += 32) {
      v16h aat, bvt;
#pragma unroll
      for (int h = 0; h < 16; ++h) {
        int K = (h & 7) + 8 * (hiL + 2 * (h >> 3));
        aat[h] = ATT[mrow][kk + K];
        bvt[h] = (_Float16)v[((size_t)b * C_ + wid * 16 + mrow) * N_ + ch * CK + kk + K];
      }
      oacc = __builtin_amdgcn_wmma_f32_16x16x32_f16(false, aat, false, bvt,
                                                    (short)0, oacc, false, false);
    }
    __syncthreads();
  }
#pragma unroll
  for (int r = 0; r < 8; ++r) {
    int qrow = qbase + r + 8 * hiL;
    o16[((size_t)b * C_ + wid * 16 + mrow) * N_ + qrow] = oacc[r];
  }
}

// -------------------- per-batch top-k via bitonic sort --------------------
__global__ __launch_bounds__(1024) void topk_kernel(const float* __restrict__ imp,
                                                    int* __restrict__ idxout,
                                                    float* __restrict__ maskbuf) {
  __shared__ float val[4096];
  __shared__ int   key[4096];
  int b = blockIdx.x, t = threadIdx.x;
  for (int i = t; i < 4096; i += 1024) {
    val[i] = (i < N_) ? imp[(size_t)b * N_ + i] : -1e30f;
    key[i] = i;
  }
  __syncthreads();
  for (int k2 = 2; k2 <= 4096; k2 <<= 1) {
    for (int j = k2 >> 1; j > 0; j >>= 1) {
      for (int i = t; i < 4096; i += 1024) {
        int ixj = i ^ j;
        if (ixj > i) {
          bool up = ((i & k2) == 0);
          bool sw = up ? (val[i] < val[ixj]) : (val[i] > val[ixj]);
          if (sw) {
            float tv = val[i]; val[i] = val[ixj]; val[ixj] = tv;
            int   tk = key[i]; key[i] = key[ixj]; key[ixj] = tk;
          }
        }
      }
      __syncthreads();
    }
  }
  for (int i = t; i < KPAD; i += 1024) idxout[b * KPAD + i] = (i < KSEL) ? key[i] : key[0];
  for (int i = t; i < N_; i += 1024) maskbuf[(size_t)b * N_ + i] = 0.f;
  __syncthreads();
  for (int i = t; i < KSEL; i += 1024) maskbuf[(size_t)b * N_ + key[i]] = 1.f;
}

// -------------------- fp32 top-k refinement (gathered attention) --------------------
__global__ __launch_bounds__(256) void refine_kernel(
    const float* __restrict__ q, const float* __restrict__ k, const float* __restrict__ v,
    const int* __restrict__ idx, float* __restrict__ o32) {
  __shared__ float SR[16][KPAD];
  __shared__ float rm[16], rs[16];
  __shared__ int   qi[16];
  __shared__ int   kidx[KPAD];

  int bx    = blockIdx.x;
  int b     = bx / QT_;
  int qbase = (bx % QT_) * 16;
  int t     = threadIdx.x;

  for (int j = t; j < KSEL; j += 256) kidx[j] = idx[b * KPAD + j];
  if (t < 16) {
    int g = qbase + t;
    qi[t] = idx[b * KPAD + ((g < KSEL) ? g : 0)];
  }
  __syncthreads();

  int i = t >> 4;  // row 0..15
  for (int j = t & 15; j < KSEL; j += 16) {
    int kj = kidx[j];
    float s = 0.f;
    const float* qp = q + ((size_t)(b * N_ + qi[i])) * CQK_;
#pragma unroll
    for (int c = 0; c < CQK_; ++c)
      s += qp[c] * k[((size_t)b * CQK_ + c) * N_ + kj];
    SR[i][j] = s;
  }
  __syncthreads();
  if (t < 16) {
    float m = -1e30f;
    for (int j = 0; j < KSEL; ++j) m = fmaxf(m, SR[t][j]);
    float s = 0.f;
    for (int j = 0; j < KSEL; ++j) s += __expf(SR[t][j] - m);
    rm[t] = m; rs[t] = s;
  }
  __syncthreads();
  for (int j = t & 15; j < KSEL; j += 16)
    SR[i][j] = __expf(SR[i][j] - rm[i]) / rs[i];
  __syncthreads();
  if (qbase + i < KSEL) {
    for (int c = t & 15; c < C_; c += 16) {
      float acc = 0.f;
      const float* vp = v + ((size_t)b * C_ + c) * N_;
      for (int j = 0; j < KSEL; ++j) acc += SR[i][j] * vp[kidx[j]];
      o32[((size_t)b * C_ + c) * N_ + qi[i]] = acc;
    }
  }
}

// -------------------- residual + gamma --------------------
__global__ __launch_bounds__(256) void residual_kernel(
    const float* __restrict__ o16, const float* __restrict__ o32,
    const float* __restrict__ x, const float* __restrict__ gamma,
    float* __restrict__ b16, float* __restrict__ b32) {
  int i = blockIdx.x * blockDim.x + threadIdx.x;
  if (i >= B_ * C_ * N_) return;
  float g = gamma[0];
  float xv = x[i];
  float o16r = (float)(_Float16)o16[i];               // out16 lives in fp16 domain
  b16[i] = g * o16r + (float)(_Float16)xv;
  b32[i] = g * o32[i] + xv;
}

// -------------------- pack outputs: s16, s32, mask --------------------
__global__ __launch_bounds__(256) void pack_kernel(
    const float* __restrict__ s16, const float* __restrict__ s32,
    const float* __restrict__ maskbuf, float* __restrict__ out) {
  int i = blockIdx.x * blockDim.x + threadIdx.x;
  const int n1 = B_ * C_ * N_;
  const int total = 2 * n1 + B_ * N_;
  if (i >= total) return;
  if (i < n1)            out[i] = s16[i];
  else if (i < 2 * n1)   out[i] = s32[i - n1];
  else                   out[i] = maskbuf[i - 2 * n1];
}

// ==================== host launcher ====================
extern "C" void kernel_launch(void* const* d_in, const int* in_sizes, int n_in,
                              void* d_out, int out_size, void* d_ws, size_t ws_size,
                              hipStream_t stream) {
  (void)in_sizes; (void)n_in; (void)out_size; (void)ws_size;
  float* ws   = (float*)d_ws;
  float* pos  = ws + OFF_POS;
  float* xbuf = ws + OFF_X;
  float* qbuf = ws + OFF_Q;
  float* kbuf = ws + OFF_K;
  float* vbuf = ws + OFF_V;
  float* o16  = ws + OFF_O16;
  float* o32  = ws + OFF_O32;
  float* impb = ws + OFF_IMP;
  int*   idxb = (int*)(ws + OFF_IDX);
  float* mask = ws + OFF_MASK;
  float* b16  = ws + OFF_B16;
  float* b32  = ws + OFF_B32;
  float* s16  = ws + OFF_S16;
  float* s32  = ws + OFF_S32;

  const int ELT   = B_ * C_ * N_;
  const int GELT  = (ELT + 255) / 256;
  const int GCONV = B_ * (N_ / 16);     // implicit-GEMM conv tiles

  pos_kernel<<<(C_ * N_ + 255) / 256, 256, 0, stream>>>(pos);

  for (int L = 0; L < 2; ++L) {
    const int base = 1 + 11 * L;
    const float* pre_w  = (const float*)d_in[base + 0];
    const float* pre_bn = (const float*)d_in[base + 1];
    const float* qw     = (const float*)d_in[base + 2];
    const float* qb     = (const float*)d_in[base + 3];
    const float* kw     = (const float*)d_in[base + 4];
    const float* kb     = (const float*)d_in[base + 5];
    const float* vw     = (const float*)d_in[base + 6];
    const float* vb     = (const float*)d_in[base + 7];
    const float* fin_w  = (const float*)d_in[base + 8];
    const float* fin_bn = (const float*)d_in[base + 9];
    const float* gamma  = (const float*)d_in[base + 10];

    const float* xin = (L == 0) ? (const float*)d_in[0] : s16;
    const int    cin = (L == 0) ? CIN0 : C_;

    conv_bn_kernel<<<GCONV, 128, 0, stream>>>(xin, cin, pre_w, pre_bn, pos, xbuf, 0);
    qkv_kernel<<<(B_ * N_ * 96 + 255) / 256, 256, 0, stream>>>(
        xbuf, qw, qb, kw, kb, vw, vb, qbuf, kbuf, vbuf);
    zero_kernel<<<(B_ * N_ + 255) / 256, 256, 0, stream>>>(impb, B_ * N_);
    attn_kernel<<<B_ * (N_ / 16), 128, 0, stream>>>(qbuf, kbuf, vbuf, o16, impb);
    topk_kernel<<<B_, 1024, 0, stream>>>(impb, idxb, mask);
    copy_kernel<<<GELT, 256, 0, stream>>>(o32, o16, ELT);
    refine_kernel<<<B_ * QT_, 256, 0, stream>>>(qbuf, kbuf, vbuf, idxb, o32);
    residual_kernel<<<GELT, 256, 0, stream>>>(o16, o32, xbuf, gamma, b16, b32);
    conv_bn_kernel<<<GCONV, 128, 0, stream>>>(b16, C_, fin_w, fin_bn, nullptr, s16, 1);
    conv_bn_kernel<<<GCONV, 128, 0, stream>>>(b32, C_, fin_w, fin_bn, nullptr, s32, 0);
  }

  const int total = 2 * ELT + B_ * N_;
  pack_kernel<<<(total + 255) / 256, 256, 0, stream>>>(s16, s32, mask, (float*)d_out);
}
